// GCNBlock_18743237280160
// MI455X (gfx1250) — matrix operand order repaired and checked by reading, not compiled, not run on hardware
//
#include <hip/hip_runtime.h>

// ---------------- problem constants ----------------
#define NB   128   // batch
#define CC   64    // in channels
#define TT   256   // time
#define VV   25    // vertices
#define SS   3     // subsets
#define OO   64    // out channels
#define RR   8     // rel channels
#define TCH  16    // t-chunk per workgroup
#define NCH  (TT/TCH)
#define MROW 26          // (t,v) row pitch: 25 real + 1 zero pad
#define VP   32          // padded V (K dim of GEMM2 / u tiles)
#define NTVF 819200.0f   // N*T*V (BN population per channel)

typedef __attribute__((ext_vector_type(8)))  float        v8f;
typedef __attribute__((ext_vector_type(8)))  unsigned int v8u;
typedef __attribute__((ext_vector_type(16))) __bf16       v16bf;
typedef unsigned short u16;

__device__ __forceinline__ u16 f2bf(float f) {
  unsigned int u = __builtin_bit_cast(unsigned int, f);
  u = (u + 0x7FFFu + ((u >> 16) & 1u)) >> 16;   // round-to-nearest-even
  return (u16)u;
}

// ---------------- kernel 1: xm = mean over T ----------------
// one block per (n,c); 6400 contiguous floats; LDS ds_add_f32 accumulation
__global__ void __launch_bounds__(256) kmean(const float* __restrict__ x,
                                             float* __restrict__ xm) {
  __shared__ float acc[VV];
  const int nc  = blockIdx.x;
  const int tid = threadIdx.x;
  if (tid < VV) acc[tid] = 0.f;
  __syncthreads();
  const float* p = x + (size_t)nc * (TT * VV);
  for (int i = tid; i < TT * VV; i += 256)
    atomicAdd(&acc[i % VV], p[i]);
  __syncthreads();
  if (tid < VV) xm[nc * VV + tid] = acc[tid] * (1.0f / (float)TT);
}

// ---------------- kernel 2: adaptive adjacency A ---------------------------
// output padded bf16, layout [s][n][o][u=32][v=32], zeros in pads
__global__ void __launch_bounds__(256) kaff(
    const float* __restrict__ xm,
    const float* __restrict__ w1, const float* __restrict__ b1,
    const float* __restrict__ w2, const float* __restrict__ b2,
    const float* __restrict__ w3, const float* __restrict__ b3,
    const float* __restrict__ PA, const float* __restrict__ alpha,
    u16* __restrict__ Abuf) {
  __shared__ float xs[CC * VV];
  __shared__ float x1s[RR * VV];
  __shared__ float x2s[RR * VV];
  __shared__ float affs[RR * VV * VV];
  const int s = blockIdx.x / NB;
  const int n = blockIdx.x % NB;
  const int tid = threadIdx.x;
  const float alphaV = alpha[0];

  for (int i = tid; i < CC * VV; i += 256) xs[i] = xm[(size_t)n * CC * VV + i];
  __syncthreads();

  if (tid < RR * VV) {
    const int r = tid / VV, v = tid % VV;
    float a1 = b1[s * RR + r], a2 = b2[s * RR + r];
    for (int c = 0; c < CC; ++c) {
      const float xv = xs[c * VV + v];
      a1 += xv * w1[(s * RR + r) * CC + c];
      a2 += xv * w2[(s * RR + r) * CC + c];
    }
    x1s[tid] = a1; x2s[tid] = a2;
  }
  __syncthreads();

  for (int i = tid; i < RR * VV * VV; i += 256) {
    const int r = i / (VV * VV), u = (i / VV) % VV, v = i % VV;
    affs[i] = tanhf(x1s[r * VV + u] - x2s[r * VV + v]);
  }
  __syncthreads();

  u16* outp = Abuf + ((size_t)(s * NB + n)) * OO * VP * VP;
  for (int i = tid; i < OO * VP * VP; i += 256) {
    const int o = i >> 10, u = (i >> 5) & 31, v = i & 31;
    float val = 0.0f;
    if (u < VV && v < VV) {
      float a = b3[s * OO + o];
#pragma unroll
      for (int r = 0; r < RR; ++r)
        a += affs[r * VV * VV + u * VV + v] * w3[(s * OO + o) * RR + r];
      val = a * alphaV + PA[(s * VV + u) * VV + v];
    }
    outp[i] = f2bf(val);
  }
}

// ---------------- kernel 3: fused r = wres*x, z += A*r, BN stats ------------
// LDS: XT[416][64] | WR[64][64] | RT[16][32][64] | AT[64][32][32]  (all bf16)
#define OFF_XT 0
#define OFF_WR 53248
#define OFF_RT 61440
#define OFF_AT 126976
#define LDS_BYTES 258048
#define NTHR 512          // 16 waves; 4 o-channels per wave

__global__ void __launch_bounds__(NTHR, 1) kmain(
    const float* __restrict__ x,
    const float* __restrict__ wres, const float* __restrict__ bres,
    const u16* __restrict__ Abuf,
    float* __restrict__ ybuf, float* __restrict__ stats) {
  extern __shared__ char smem[];
  u16* XT = (u16*)(smem + OFF_XT);
  u16* WR = (u16*)(smem + OFF_WR);
  u16* RT = (u16*)(smem + OFF_RT);
  u16* AT = (u16*)(smem + OFF_AT);

  const int tchunk = blockIdx.x;
  const int n      = blockIdx.y;
  const int t0     = tchunk * TCH;
  const int tid    = threadIdx.x;
  const int wave   = tid >> 5;
  const int lane   = tid & 31;
  const int lcol   = lane & 15;
  const int khA    = (lane < 16) ? 0 : 8;    // A-matrix K-half per lane group
  const int kbB    = (lane < 16) ? 0 : 16;   // unused; B operands index k with khA as A does
  const int mrowB  = (lane < 16) ? 0 : 8;    // D-matrix row base per lane group

  // ---- stage x tile: XT[m=(t*26+v)][c] bf16, pad rows zero ----
  const float* xp = x + (((size_t)n * CC) * TT + t0) * VV;  // (n, c=0, t0, 0)
  for (int i = tid; i < CC * (TCH * VV); i += NTHR) {
    const int c = i / (TCH * VV), j = i % (TCH * VV);
    const int t = j / VV, v = j % VV;
    XT[(t * MROW + v) * CC + c] = f2bf(xp[(size_t)c * (TT * VV) + j]);
  }
  for (int i = tid; i < TCH * CC; i += NTHR) {           // pad row v==25
    const int t = i >> 6, c = i & 63;
    XT[(t * MROW + 25) * CC + c] = 0;
  }
  for (int i = tid; i < TCH * (VP - VV) * OO; i += NTHR) { // RT K-pad v=25..31
    const int t = i / ((VP - VV) * OO);
    const int rem = i % ((VP - VV) * OO);
    const int v = VV + rem / OO, o = rem & 63;
    RT[(t * VP + v) * OO + o] = 0;
  }

  // persistent z accumulators: 4 o-channels per wave, 2 u-tiles each (64 VGPRs)
  v8f zacc[4][2];
#pragma unroll
  for (int q = 0; q < 4; ++q)
#pragma unroll
    for (int ut = 0; ut < 2; ++ut) zacc[q][ut] = (v8f)0.0f;

  for (int s = 0; s < SS; ++s) {
    __syncthreads();  // prev GEMM2 done before WR/AT overwrite
    // ---- stage wres (B: [k=c][n=o]) and padded A tile ----
    const float* wrp = wres + (size_t)s * OO * CC;
    for (int i = tid; i < CC * OO; i += NTHR) {
      const int c = i >> 6, o = i & 63;
      WR[c * OO + o] = f2bf(wrp[o * CC + c]);
    }
    const u16* ap = Abuf + ((size_t)(s * NB + n)) * OO * VP * VP;
    for (int i = tid; i < OO * VP * VP; i += NTHR) AT[i] = ap[i];
    __syncthreads();

    // ---- GEMM1: r[(t,v), o] = sum_c x[(t,v),c] * wres[o,c]  (26x4 tiles, K=64)
    for (int job = wave; job < 26 * 4; job += 16) {
      const int mt = job >> 2, nt = job & 3;
      const int m = mt * 16 + lcol;
      const int ocol = nt * 16 + lcol;
      v8f acc = (v8f)0.0f;
#pragma unroll
      for (int kk = 0; kk < 2; ++kk) {
        v8u a, b;
#pragma unroll
        for (int i = 0; i < 8; ++i) {
          const int k2 = (i < 4) ? (2 * i + khA) : (16 + 2 * (i - 4) + khA);
          a[i] = *(const unsigned int*)&XT[m * CC + kk * 32 + k2];
          const int c = kk * 32 + k2;
          const unsigned lo = WR[c * OO + ocol];
          const unsigned hi = WR[(c + 1) * OO + ocol];
          b[i] = lo | (hi << 16);
        }
        acc = __builtin_amdgcn_wmma_f32_16x16x32_bf16(
            false, __builtin_bit_cast(v16bf, a),
            false, __builtin_bit_cast(v16bf, b),
            (short)0, acc, false, false);
      }
      const float bias = bres[s * OO + ocol];
#pragma unroll
      for (int i = 0; i < 8; ++i) {
        const int mm = mt * 16 + i + mrowB;
        const int t = mm / MROW, v = mm % MROW;
        if (v < VV) RT[(t * VP + v) * OO + ocol] = f2bf(acc[i] + bias);
      }
    }
    __syncthreads();

    // ---- GEMM2: z[t,u] += sum_v r[t,v] * A[u,v]   (per o, K = v padded to 32)
#pragma unroll
    for (int q = 0; q < 4; ++q) {
      const int o = wave * 4 + q;
      v8u a;
#pragma unroll
      for (int i = 0; i < 8; ++i) {
        const int k2 = (i < 4) ? (2 * i + khA) : (16 + 2 * (i - 4) + khA);
        const unsigned lo = RT[(lcol * VP + k2) * OO + o];
        const unsigned hi = RT[(lcol * VP + k2 + 1) * OO + o];
        a[i] = lo | (hi << 16);
      }
#pragma unroll
      for (int ut = 0; ut < 2; ++ut) {
        const int u = ut * 16 + lcol;
        v8u b;
#pragma unroll
        for (int i = 0; i < 8; ++i) {
          const int v0 = (i < 4) ? (2 * i + khA) : (16 + 2 * (i - 4) + khA);
          // padded AT: single aligned 4-byte read of (v0, v0+1) pair
          b[i] = *(const unsigned int*)&AT[(o * VP + u) * VP + v0];
        }
        zacc[q][ut] = __builtin_amdgcn_wmma_f32_16x16x32_bf16(
            false, __builtin_bit_cast(v16bf, a),
            false, __builtin_bit_cast(v16bf, b),
            (short)0, zacc[q][ut], false, false);
      }
    }
  }

  // ---- epilogue: write y, accumulate BN stats (fused stats pass) ----
#pragma unroll
  for (int q = 0; q < 4; ++q) {
    const int o = wave * 4 + q;
    float ssum = 0.f, ssq = 0.f;
#pragma unroll
    for (int ut = 0; ut < 2; ++ut) {
      const int u = ut * 16 + lcol;
#pragma unroll
      for (int i = 0; i < 8; ++i) {
        const float val = zacc[q][ut][i];
        const int t = t0 + i + mrowB;
        if (u < VV) {
          ybuf[(((size_t)n * OO + o) * TT + t) * VV + u] = val;
          ssum += val; ssq += val * val;
        }
      }
    }
    for (int off = 16; off > 0; off >>= 1) {
      ssum += __shfl_down(ssum, off, 32);
      ssq  += __shfl_down(ssq,  off, 32);
    }
    if (lane == 0) {
      atomicAdd(&stats[o],      ssum);
      atomicAdd(&stats[OO + o], ssq);
    }
  }
}

// ---------------- kernel 4: BN normalize + residual + ReLU ------------------
__global__ void __launch_bounds__(256) kfinal(
    const float* __restrict__ ybuf, const float* __restrict__ x,
    const float* __restrict__ stats,
    const float* __restrict__ gamma, const float* __restrict__ beta,
    float* __restrict__ out) {
  const size_t idx4 = (size_t)blockIdx.x * blockDim.x + threadIdx.x;
  const size_t base = idx4 * 4;
  const int o = (int)((base / (TT * VV)) & (OO - 1));
  const float inv = 1.0f / NTVF;
  const float m  = stats[o] * inv;
  const float vr = fmaxf(stats[OO + o] * inv - m * m, 0.0f);
  const float sc = gamma[o] * rsqrtf(vr + 1e-5f);
  const float bt = beta[o] - m * sc;
  const float4 yv = ((const float4*)ybuf)[idx4];
  const float4 xv = ((const float4*)x)[idx4];
  float4 r;
  r.x = fmaxf(yv.x * sc + bt + xv.x, 0.0f);
  r.y = fmaxf(yv.y * sc + bt + xv.y, 0.0f);
  r.z = fmaxf(yv.z * sc + bt + xv.z, 0.0f);
  r.w = fmaxf(yv.w * sc + bt + xv.w, 0.0f);
  ((float4*)out)[idx4] = r;
}

// ---------------- launch ----------------------------------------------------
extern "C" void kernel_launch(void* const* d_in, const int* in_sizes, int n_in,
                              void* d_out, int out_size, void* d_ws, size_t ws_size,
                              hipStream_t stream) {
  const float* x     = (const float*)d_in[0];
  const float* w1    = (const float*)d_in[1];
  const float* b1    = (const float*)d_in[2];
  const float* w2    = (const float*)d_in[3];
  const float* b2    = (const float*)d_in[4];
  const float* w3    = (const float*)d_in[5];
  const float* b3    = (const float*)d_in[6];
  const float* wres  = (const float*)d_in[7];
  const float* bres  = (const float*)d_in[8];
  const float* PA    = (const float*)d_in[9];
  const float* alpha = (const float*)d_in[10];
  const float* gamma = (const float*)d_in[11];
  const float* beta  = (const float*)d_in[12];
  float* out = (float*)d_out;

  // workspace layout (bytes)
  const size_t xm_off    = 0;                 // 128*64*25*4        =     819,200
  const size_t abuf_off  = 819200;            // 3*128*64*32*32*2   =  50,331,648
  const size_t ybuf_off  = 51150848;          // 128*64*256*25*4    = 209,715,200
  const size_t stats_off = 260866048;         // 2*64*4             =         512
  if (ws_size < stats_off + 512) return;

  char* ws = (char*)d_ws;
  float* xm    = (float*)(ws + xm_off);
  u16*   Abuf  = (u16*)  (ws + abuf_off);
  float* ybuf  = (float*)(ws + ybuf_off);
  float* stats = (float*)(ws + stats_off);

  hipMemsetAsync(stats, 0, 2 * OO * sizeof(float), stream);
  kmean<<<NB * CC, 256, 0, stream>>>(x, xm);
  kaff<<<SS * NB, 256, 0, stream>>>(xm, w1, b1, w2, b2, w3, b3, PA, alpha, Abuf);
  kmain<<<dim3(NCH, NB), NTHR, LDS_BYTES, stream>>>(x, wres, bres, Abuf, ybuf, stats);
  kfinal<<<(NB * CC * TT * VV) / (256 * 4), 256, 0, stream>>>(ybuf, x, stats, gamma, beta, out);
}
